// MultiHeadAttention_15204184228092
// MI455X (gfx1250) — matrix operand order repaired and checked by reading, not compiled, not run on hardware
//
#include <hip/hip_runtime.h>
#include <hip/hip_fp16.h>
#include <stdint.h>

#define D_MODEL 512
#define NHEAD   8
#define HDIM    64
#define BATCH   8
#define SEQ     1024
#define NTOK    (BATCH * SEQ)   // 8192

typedef __attribute__((ext_vector_type(16))) _Float16 v16h;
typedef __attribute__((ext_vector_type(8)))  float    v8f;

union Frag {
  v16h     v;
  _Float16 h[16];
  uint4    q[2];
};

__device__ __forceinline__ v8f wmma_f16(const Frag &a, const Frag &b, v8f c) {
  return __builtin_amdgcn_wmma_f32_16x16x32_f16(false, a.v, false, b.v,
                                                (short)0, c, false, false);
}

// A operand (16 rows x 32 K), source row-major f16 with leading dim lda.
// ISA layout: lanes 0-15 row M=lane, halves 0-7 = K 0..7, halves 8-15 = K 16..23;
// lanes 16-31 same rows, K shifted by 8.
__device__ __forceinline__ Frag load_a(const _Float16 *A, int lda, int lane) {
  Frag f;
  const int row = lane & 15;
  const int kb  = (lane >> 4) * 8;
  const _Float16 *p = A + row * lda;
  f.q[0] = *(const uint4 *)(p + kb);        // K kb..kb+7
  f.q[1] = *(const uint4 *)(p + kb + 16);   // K kb+16..kb+23
  return f;
}

// B operand (32 K x 16 N) read from a TRANSPOSED matrix Bt[N][K] (row-major),
// i.e. B[k][n] = Bt[n][k].  Lane: col = lane&15, K base = (lane>>4)*8, halves 8-15 at K base+16.
// -> two 16-byte reads per lane.
__device__ __forceinline__ Frag load_bt(const _Float16 *Bt, int ldb, int lane) {
  Frag f;
  const int col = lane & 15;
  const int kb  = (lane >> 4) * 8;
  const _Float16 *p = Bt + col * ldb + kb;
  f.q[0] = *(const uint4 *)(p);
  f.q[1] = *(const uint4 *)(p + 16);
  return f;
}

// ---------------- conversion kernels ----------------

__global__ void __launch_bounds__(256) cvt_f16_kernel(const float *__restrict__ in,
                                                      _Float16 *__restrict__ out,
                                                      int n) {
  int i = (blockIdx.x * blockDim.x + threadIdx.x) * 4;
  if (i + 3 < n) {
    float4 v = *(const float4 *)(in + i);
    out[i + 0] = (_Float16)v.x;
    out[i + 1] = (_Float16)v.y;
    out[i + 2] = (_Float16)v.z;
    out[i + 3] = (_Float16)v.w;
  }
}

// in: [K, N] fp32  ->  out: [N, K] f16 (transposed)
__global__ void __launch_bounds__(256) cvt_transpose_kernel(const float *__restrict__ in,
                                                            _Float16 *__restrict__ out,
                                                            int K, int N) {
  int nn = blockIdx.x * 16 + threadIdx.x;
  int kk = blockIdx.y * 16 + threadIdx.y;
  if (nn < N && kk < K) out[nn * K + kk] = (_Float16)in[kk * N + nn];
}

// ---------------- QKV projection GEMM ----------------
// [8192,512]f16 @ [512,1536] (as WT[1536,512]) + bias -> scatter Q/K/Vt f16
// Double-buffered K loop: next chunk's 5 fragments load while current 4 WMMAs run.

__global__ void __launch_bounds__(128) qkv_gemm_kernel(
    const _Float16 *__restrict__ Xh,     // [NTOK, 512]
    const _Float16 *__restrict__ WT,     // [1536, 512] (transposed weights)
    const float    *__restrict__ bias,   // [1536]
    _Float16 *__restrict__ Qh,           // [B,H,L,64]  (pre-scaled by 1/8)
    _Float16 *__restrict__ Kh,           // [B,H,L,64]
    _Float16 *__restrict__ Vt) {         // [B,H,64,L]  (transposed V)
  const int lane = threadIdx.x & 31;
  const int wave = threadIdx.x >> 5;
  const int n0 = blockIdx.x * 64;
  const int m0 = blockIdx.y * 64 + wave * 16;

  const _Float16 *Ap = Xh + (size_t)m0 * D_MODEL;

  v8f zero = {};
  v8f acc[4];
#pragma unroll
  for (int j = 0; j < 4; ++j) acc[j] = zero;

  Frag a_cur = load_a(Ap, D_MODEL, lane);
  Frag b_cur[4];
#pragma unroll
  for (int j = 0; j < 4; ++j)
    b_cur[j] = load_bt(WT + (size_t)(n0 + j * 16) * D_MODEL, D_MODEL, lane);

  for (int kc = 0; kc < D_MODEL; kc += 32) {
    const int kn = (kc + 32 < D_MODEL) ? kc + 32 : 0;  // branchless prefetch
    Frag a_nxt = load_a(Ap + kn, D_MODEL, lane);
    Frag b_nxt[4];
#pragma unroll
    for (int j = 0; j < 4; ++j)
      b_nxt[j] = load_bt(WT + (size_t)(n0 + j * 16) * D_MODEL + kn, D_MODEL, lane);

#pragma unroll
    for (int j = 0; j < 4; ++j) acc[j] = wmma_f16(a_cur, b_cur[j], acc[j]);

    a_cur = a_nxt;
#pragma unroll
    for (int j = 0; j < 4; ++j) b_cur[j] = b_nxt[j];
  }

  const int hi  = lane >> 4;
  const int col = lane & 15;
#pragma unroll
  for (int j = 0; j < 4; ++j) {
    int n    = n0 + j * 16 + col;
    int head = n / 192;
    int w    = n % 192;
    int typ  = w / 64;
    int d    = w % 64;
    float bv = bias[n];
#pragma unroll
    for (int r = 0; r < 8; ++r) {
      int tok = m0 + r + hi * 8;
      int b   = tok >> 10;
      int l   = tok & 1023;
      float val = acc[j][r] + bv;
      if (typ == 0)
        Qh[((b * NHEAD + head) * SEQ + l) * HDIM + d] = (_Float16)(val * 0.125f);
      else if (typ == 1)
        Kh[((b * NHEAD + head) * SEQ + l) * HDIM + d] = (_Float16)val;
      else
        Vt[((b * NHEAD + head) * HDIM + d) * SEQ + l] = (_Float16)val;
    }
  }
}

// ---------------- flash attention ----------------
// grid (SEQ/64, NHEAD, BATCH), 4 waves/block, each wave: 16 q-rows x 64 dims.

__global__ void __launch_bounds__(128) attn_kernel(
    const _Float16 *__restrict__ Qh,   // [B,H,L,64]
    const _Float16 *__restrict__ Kh,   // [B,H,L,64]
    const _Float16 *__restrict__ Vt,   // [B,H,64,L]
    const float    *__restrict__ mask, // [L,L]
    _Float16 *__restrict__ Oh) {       // [NTOK, 512] context (f16)
  __shared__ __align__(16) _Float16 Plds[4][16 * 32];

  const int lane = threadIdx.x & 31;
  const int wave = threadIdx.x >> 5;
  const int q0   = blockIdx.x * 64 + wave * 16;
  const int h    = blockIdx.y;
  const int b    = blockIdx.z;
  const int bh   = b * NHEAD + h;
  const int hi   = lane >> 4;
  const int col  = lane & 15;

  const _Float16 *Qp = Qh + ((size_t)bh * SEQ + q0) * HDIM;
  const _Float16 *Kp = Kh + (size_t)bh * SEQ * HDIM;
  const _Float16 *Vp = Vt + (size_t)bh * HDIM * SEQ;
  _Float16 *myP = &Plds[wave][0];

  Frag aq[2];
  aq[0] = load_a(Qp + 0, HDIM, lane);
  aq[1] = load_a(Qp + 32, HDIM, lane);

  v8f zero = {};
  v8f o[4];
#pragma unroll
  for (int j = 0; j < 4; ++j) o[j] = zero;
  float mrow[8], lrow[8];
#pragma unroll
  for (int r = 0; r < 8; ++r) { mrow[r] = -3.0e38f; lrow[r] = 0.0f; }

  for (int kt = 0; kt < SEQ; kt += 32) {
    if (kt + 32 < SEQ) {
      __builtin_prefetch(Kp + (kt + 32) * HDIM, 0, 0);
      __builtin_prefetch(Vp + kt + 32, 0, 0);
    }
    // ---- batch-load K fragments, then S = (Q/8) @ K^T (2 tiles of 16) ----
    Frag bk[2][2];
#pragma unroll
    for (int t = 0; t < 2; ++t)
#pragma unroll
      for (int c = 0; c < 2; ++c)
        bk[t][c] = load_bt(Kp + (kt + t * 16) * HDIM + c * 32, HDIM, lane);

    v8f s[2];
#pragma unroll
    for (int t = 0; t < 2; ++t) {
      v8f sa = zero;
      sa = wmma_f16(aq[0], bk[t][0], sa);
      sa = wmma_f16(aq[1], bk[t][1], sa);
      s[t] = sa;
    }

    // ---- issue V fragment loads now; latency hidden under softmax VALU ----
    Frag bv[4];
#pragma unroll
    for (int j = 0; j < 4; ++j)
      bv[j] = load_bt(Vp + (j * 16) * SEQ + kt, SEQ, lane);

    // ---- + mask ----
#pragma unroll
    for (int r = 0; r < 8; ++r) {
      int qrow = q0 + r + hi * 8;
      const float *mp = mask + (size_t)qrow * SEQ + kt + col;
      s[0][r] += mp[0];
      s[1][r] += mp[16];
    }
    // ---- online softmax (row reductions across 16-lane groups) ----
#pragma unroll
    for (int r = 0; r < 8; ++r) {
      float rm = fmaxf(s[0][r], s[1][r]);
      rm = fmaxf(rm, __shfl_xor(rm, 1, 32));
      rm = fmaxf(rm, __shfl_xor(rm, 2, 32));
      rm = fmaxf(rm, __shfl_xor(rm, 4, 32));
      rm = fmaxf(rm, __shfl_xor(rm, 8, 32));
      float mnew = fmaxf(mrow[r], rm);
      float sc   = __expf(mrow[r] - mnew);
      mrow[r] = mnew;
      lrow[r] *= sc;
      o[0][r] *= sc; o[1][r] *= sc; o[2][r] *= sc; o[3][r] *= sc;

      float p0 = __expf(s[0][r] - mnew);
      float p1 = __expf(s[1][r] - mnew);
      int prow = r + hi * 8;
      myP[prow * 32 + col]      = (_Float16)p0;   // keys kt..kt+15
      myP[prow * 32 + 16 + col] = (_Float16)p1;   // keys kt+16..kt+31
      float ps = p0 + p1;
      ps += __shfl_xor(ps, 1, 32);
      ps += __shfl_xor(ps, 2, 32);
      ps += __shfl_xor(ps, 4, 32);
      ps += __shfl_xor(ps, 8, 32);
      lrow[r] += ps;
    }
    // wave-private LDS transpose: accumulator layout -> A layout
    asm volatile("s_wait_dscnt 0" ::: "memory");
    Frag ap;
    {
      const int row = lane & 15;
      const int kb  = hi * 8;
      ap.q[0] = *(const uint4 *)(myP + row * 32 + kb);
      ap.q[1] = *(const uint4 *)(myP + row * 32 + kb + 16);
    }
    // ---- O += P @ V ----
#pragma unroll
    for (int j = 0; j < 4; ++j) o[j] = wmma_f16(ap, bv[j], o[j]);
  }

  // ---- normalize + store context (f16, [B,L, h*64+d]) ----
#pragma unroll
  for (int r = 0; r < 8; ++r) {
    float inv = 1.0f / lrow[r];
    int qrow  = q0 + r + hi * 8;
    _Float16 *op = Oh + ((size_t)(b * SEQ + qrow)) * D_MODEL + h * HDIM + col;
    op[0]  = (_Float16)(o[0][r] * inv);
    op[16] = (_Float16)(o[1][r] * inv);
    op[32] = (_Float16)(o[2][r] * inv);
    op[48] = (_Float16)(o[3][r] * inv);
  }
}

// ---------------- output projection GEMM ----------------
// [8192,512]f16 @ Wo (as WoT[512,512]) + bo -> fp32 d_out

__global__ void __launch_bounds__(128) out_gemm_kernel(
    const _Float16 *__restrict__ Ah,    // [NTOK, 512] context
    const _Float16 *__restrict__ WoT,   // [512, 512] transposed
    const float    *__restrict__ bo,    // [512]
    float *__restrict__ out) {          // [NTOK, 512]
  const int lane = threadIdx.x & 31;
  const int wave = threadIdx.x >> 5;
  const int n0 = blockIdx.x * 64;
  const int m0 = blockIdx.y * 64 + wave * 16;

  const _Float16 *Ap = Ah + (size_t)m0 * D_MODEL;

  v8f zero = {};
  v8f acc[4];
#pragma unroll
  for (int j = 0; j < 4; ++j) acc[j] = zero;

  Frag a_cur = load_a(Ap, D_MODEL, lane);
  Frag b_cur[4];
#pragma unroll
  for (int j = 0; j < 4; ++j)
    b_cur[j] = load_bt(WoT + (size_t)(n0 + j * 16) * D_MODEL, D_MODEL, lane);

  for (int kc = 0; kc < D_MODEL; kc += 32) {
    const int kn = (kc + 32 < D_MODEL) ? kc + 32 : 0;
    Frag a_nxt = load_a(Ap + kn, D_MODEL, lane);
    Frag b_nxt[4];
#pragma unroll
    for (int j = 0; j < 4; ++j)
      b_nxt[j] = load_bt(WoT + (size_t)(n0 + j * 16) * D_MODEL + kn, D_MODEL, lane);

#pragma unroll
    for (int j = 0; j < 4; ++j) acc[j] = wmma_f16(a_cur, b_cur[j], acc[j]);

    a_cur = a_nxt;
#pragma unroll
    for (int j = 0; j < 4; ++j) b_cur[j] = b_nxt[j];
  }

  const int hi  = lane >> 4;
  const int col = lane & 15;
#pragma unroll
  for (int j = 0; j < 4; ++j) {
    int n    = n0 + j * 16 + col;
    float bv = bo[n];
#pragma unroll
    for (int r = 0; r < 8; ++r) {
      int tok = m0 + r + hi * 8;
      out[(size_t)tok * D_MODEL + n] = acc[j][r] + bv;
    }
  }
}

// ---------------- launch ----------------

extern "C" void kernel_launch(void *const *d_in, const int *in_sizes, int n_in,
                              void *d_out, int out_size, void *d_ws, size_t ws_size,
                              hipStream_t stream) {
  const float *x    = (const float *)d_in[0];
  const float *mask = (const float *)d_in[1];
  const float *Wqkv = (const float *)d_in[2];
  const float *bqkv = (const float *)d_in[3];
  const float *Wo   = (const float *)d_in[4];
  const float *bo   = (const float *)d_in[5];
  float *out = (float *)d_out;

  char *ws = (char *)d_ws;
  _Float16 *Xh    = (_Float16 *)ws; ws += (size_t)NTOK * D_MODEL * 2;       // 8 MiB
  _Float16 *WqkvT = (_Float16 *)ws; ws += (size_t)3 * D_MODEL * D_MODEL * 2;// 1.5 MiB
  _Float16 *WoT   = (_Float16 *)ws; ws += (size_t)D_MODEL * D_MODEL * 2;    // 0.5 MiB
  _Float16 *Qh    = (_Float16 *)ws; ws += (size_t)NTOK * D_MODEL * 2;       // 8 MiB
  _Float16 *Kh    = (_Float16 *)ws; ws += (size_t)NTOK * D_MODEL * 2;       // 8 MiB
  _Float16 *Vt    = (_Float16 *)ws; ws += (size_t)NTOK * D_MODEL * 2;       // 8 MiB
  _Float16 *Oh    = (_Float16 *)ws; ws += (size_t)NTOK * D_MODEL * 2;       // 8 MiB

  const int n_x = NTOK * D_MODEL;
  cvt_f16_kernel<<<n_x / 1024, 256, 0, stream>>>(x, Xh, n_x);
  cvt_transpose_kernel<<<dim3(3 * D_MODEL / 16, D_MODEL / 16), dim3(16, 16), 0, stream>>>(
      Wqkv, WqkvT, D_MODEL, 3 * D_MODEL);
  cvt_transpose_kernel<<<dim3(D_MODEL / 16, D_MODEL / 16), dim3(16, 16), 0, stream>>>(
      Wo, WoT, D_MODEL, D_MODEL);

  qkv_gemm_kernel<<<dim3(3 * D_MODEL / 64, NTOK / 64), 128, 0, stream>>>(
      Xh, WqkvT, bqkv, Qh, Kh, Vt);

  attn_kernel<<<dim3(SEQ / 64, NHEAD, BATCH), 128, 0, stream>>>(
      Qh, Kh, Vt, mask, Oh);

  out_gemm_kernel<<<dim3(D_MODEL / 64, NTOK / 64), 128, 0, stream>>>(
      Oh, WoT, bo, out);
}
